// NodeEncodeInterface_43714177138802
// MI455X (gfx1250) — compile-verified, matrix operand order checked by reading, not run on hardware
//
#include <hip/hip_runtime.h>
#include <hip/hip_bf16.h>
#include <stdint.h>

// Problem constants (match reference)
#define N_NODES 300000
#define N_EDGES 1200000
#define HID 256
#define EMB 32
#define DIN 288   // EMB + HID

// Register blocking for the WMMA GEMM: each wave computes (MT*16) x (NT*16).
// 18750 M-tiles divisible by MT=3; N-tiles (16 and 32) divisible by NT=4.
#define MT 3
#define NT 4
#define LDS_PAD 8   // elements of row padding in the LDS A panel (16B)

typedef __attribute__((ext_vector_type(16))) __bf16        v16bf;
typedef __attribute__((ext_vector_type(8)))  float         v8f;
typedef __attribute__((ext_vector_type(8)))  unsigned short v8u16;
typedef __attribute__((__vector_size__(16))) int           v4i;

__device__ __forceinline__ unsigned short f2bf(float f) {
    uint32_t u = __float_as_uint(f);
    u += 0x7FFFu + ((u >> 16) & 1u);   // round-to-nearest-even
    return (unsigned short)(u >> 16);
}
__device__ __forceinline__ float bf2f(unsigned short h) {
    return __uint_as_float(((uint32_t)h) << 16);
}

// 16B global -> LDS copy; async (ASYNCcnt-tracked) when the builtin exists.
__device__ __forceinline__ void copy16_to_lds(const unsigned short* g,
                                              unsigned short* l) {
#if __has_builtin(__builtin_amdgcn_global_load_async_to_lds_b128)
    __builtin_amdgcn_global_load_async_to_lds_b128(
        (__attribute__((address_space(1))) v4i*)(void*)g,
        (__attribute__((address_space(3))) v4i*)(void*)l,
        0, 0);
#else
    *(v8u16*)l = *(const v8u16*)g;     // global_load_b128 + ds_store_b128
#endif
}

__device__ __forceinline__ void wait_async_copies() {
#if __has_builtin(__builtin_amdgcn_global_load_async_to_lds_b128)
#if __has_builtin(__builtin_amdgcn_s_wait_asynccnt)
    __builtin_amdgcn_s_wait_asynccnt(0);
#else
    asm volatile("s_wait_asynccnt 0x0" ::: "memory");
#endif
#endif
}

// ---------------------------------------------------------------- utilities
__global__ void zero_f32(float* __restrict__ p, long n) {
    long i = (long)blockIdx.x * blockDim.x + threadIdx.x;
    long stride = (long)gridDim.x * blockDim.x;
    for (; i < n; i += stride) p[i] = 0.0f;
}

// Build A_c = bf16([c_emb[sol], x]) rows; cache sol_node.
// grid = N_NODES blocks, blockDim = 288 (one thread per feature)
__global__ void build_feats(const float* __restrict__ x,
                            const int* __restrict__ batch,
                            const int* __restrict__ solcls,
                            const float* __restrict__ c_emb,
                            unsigned short* __restrict__ A_c,
                            int* __restrict__ sol_node) {
    int i = blockIdx.x;
    int j = threadIdx.x;                  // 0..287
    int sol = solcls[batch[i]];
    float v = (j < EMB) ? c_emb[sol * EMB + j]
                        : x[(long)i * HID + (j - EMB)];
    A_c[(long)i * DIN + j] = f2bf(v);
    if (j == 0) sol_node[i] = sol;
}

// One wave per edge; lanes cooperatively scatter the 288-wide message.
// Only ~E/81 edges are valid (z uniform 0..8), so atomics are cheap.
__global__ void edge_aggregate(const int* __restrict__ src,
                               const int* __restrict__ dst,
                               const int* __restrict__ z,
                               const int* __restrict__ sol_node,
                               const float* __restrict__ x,
                               const float* __restrict__ h_emb,
                               float* __restrict__ h_sum,
                               float* __restrict__ cnt) {
    int tid  = blockIdx.x * blockDim.x + threadIdx.x;
    int e    = tid >> 5;
    int lane = tid & 31;
    if (e >= N_EDGES) return;
    int s = src[e], d = dst[e];
    if (z[s] != 5 || z[d] != 0) return;   // is_c[src] & is_h[dst]
    int sol = sol_node[d];
    atomicAdd(&h_sum[(long)s * DIN + lane], h_emb[sol * EMB + lane]);
#pragma unroll
    for (int t = 1; t < 9; ++t) {         // j = 32..287: x part
        int j = lane + t * 32;
        atomicAdd(&h_sum[(long)s * DIN + j], x[(long)d * HID + (j - EMB)]);
    }
    if (lane == 0) atomicAdd(&cnt[s], 1.0f);
}

// h_avg = h_sum / max(cnt,1) -> bf16 A_h.  grid = N_NODES, blockDim = 288
__global__ void finalize_havg(const float* __restrict__ h_sum,
                              const float* __restrict__ cnt,
                              unsigned short* __restrict__ A_h) {
    int i = blockIdx.x, j = threadIdx.x;
    float inv = 1.0f / fmaxf(cnt[i], 1.0f);
    A_h[(long)i * DIN + j] = f2bf(h_sum[(long)i * DIN + j] * inv);
}

// Repack f32 W[K][Nout] into per-lane WMMA B fragments (16-bit B 32x16 layout):
// tile (kt,nt); lane holds column n = nt*16 + lane%16; its 16 K-values are
// idx 0..7 -> k_local {0..7 | 8..15} and idx 8..15 -> k_local {16..23 | 24..31}
// depending on lane half.  Fragment stored contiguously: 32 lanes * 16 bf16.
__global__ void pack_weights(const float* __restrict__ W,
                             unsigned short* __restrict__ Bp,
                             int K, int Nout) {
    int tid = blockIdx.x * blockDim.x + threadIdx.x;
    if (tid >= K * Nout) return;
    int idx  = tid & 15;
    int lane = (tid >> 4) & 31;
    int tile = tid >> 9;
    int ntilesN = Nout >> 4;
    int nt = tile % ntilesN;
    int kt = tile / ntilesN;
    int klocal = ((idx < 8) ? idx : idx + 8) + ((lane >= 16) ? 8 : 0);
    int k = kt * 32 + klocal;
    int n = nt * 16 + (lane & 15);
    Bp[tid] = f2bf(W[(long)k * Nout + n]);
}

// bf16 WMMA GEMM, 3x4 register blocking per wave (48x64 output block,
// 12 accumulators).  The 8 waves of a block share one A panel staged in
// LDS via GLOBAL_LOAD_ASYNC_TO_LDS_B128 (ASYNCcnt), cutting A global
// traffic by n_groups (4x / 8x).  B fragments stream from global (L2-
// resident, 147-262 KB).  Per K-step per wave: 12 v_wmma from 6 ds_load
// + 8 global_load b128.
__global__ void __launch_bounds__(256)
wmma_gemm(const unsigned short* __restrict__ A,
          const unsigned short* __restrict__ Bp,
          const float* __restrict__ bias,
          unsigned short* __restrict__ Out,
          int K, int KP, int Nout, int n_groups, int relu) {
    extern __shared__ unsigned short sA[];   // rows_per_block x KP bf16

    int wib  = threadIdx.x >> 5;             // wave in block: 0..7
    int lane = threadIdx.x & 31;
    int mg_sub = wib / n_groups;             // M-group within block
    int ng     = wib % n_groups;
    int mg_per_block   = 8 / n_groups;
    int rows_per_block = mg_per_block * (MT * 16);
    long block_row = (long)blockIdx.x * rows_per_block;

    // ---- stage A panel (rows_per_block x K, row-padded to KP) into LDS ----
    {
        const unsigned short* gbase = A + block_row * K;
        int chunks_per_row = K >> 3;                   // 16B chunks per row
        int nchunks = rows_per_block * chunks_per_row;
        for (int c = threadIdx.x; c < nchunks; c += 256) {
            int r = c / chunks_per_row;
            int q = c - r * chunks_per_row;
            copy16_to_lds(gbase + (long)r * K + q * 8, sA + r * KP + q * 8);
        }
        wait_async_copies();
        __syncthreads();
    }

    // ---- main loop: A fragments from LDS, B fragments from global ----
    // A fragment: lane holds row (lane%16) of the subtile; K-groups
    // [kbase..+7] and [kbase+16..+23], kbase = kt*32 + 8*(lane/16)
    int koff = (lane >> 4) * 8;
    const unsigned short* srow =
        sA + (mg_sub * (MT * 16) + (lane & 15)) * KP + koff;
    int sstride = 16 * KP;                   // between M-subtiles (elements)

    const unsigned short* bptr = Bp + ((long)ng * (NT * 32) + lane) * 16;
    long bstride = (long)(Nout >> 4) * 512;  // elements per 32-wide K slab

    v8f acc[MT][NT];
#pragma unroll
    for (int s = 0; s < MT; ++s)
#pragma unroll
        for (int j = 0; j < NT; ++j) acc[s][j] = (v8f){};

    union Frag { v16bf v; v8u16 h[2]; };
    int kt_iters = K >> 5;
    for (int kt = 0; kt < kt_iters; ++kt) {
        Frag a[MT], b[NT];
#pragma unroll
        for (int s = 0; s < MT; ++s) {
            a[s].h[0] = *(const v8u16*)(srow + s * sstride);        // ds_load_b128
            a[s].h[1] = *(const v8u16*)(srow + s * sstride + 16);
        }
#pragma unroll
        for (int j = 0; j < NT; ++j)
            b[j].v = *(const v16bf*)(bptr + j * 512);               // 32B fragment
#pragma unroll
        for (int s = 0; s < MT; ++s)
#pragma unroll
            for (int j = 0; j < NT; ++j)
                acc[s][j] = __builtin_amdgcn_wmma_f32_16x16x32_bf16(
                    false, a[s].v, false, b[j].v, (short)0, acc[s][j], false, false);
        srow += 32;
        bptr += bstride;
    }

    // C/D layout: VGPR r -> row (r + 8*(lane/16)), col = lane%16
#pragma unroll
    for (int s = 0; s < MT; ++s) {
        long mrow = block_row + mg_sub * (MT * 16) + s * 16 + (lane >> 4) * 8;
#pragma unroll
        for (int j = 0; j < NT; ++j) {
            int n = (ng * NT + j) * 16 + (lane & 15);
            float bv = bias[n];
#pragma unroll
            for (int r = 0; r < 8; ++r) {
                float v = acc[s][j][r] + bv;
                if (relu) v = fmaxf(v, 0.0f);
                Out[(mrow + r) * Nout + n] = f2bf(v);
            }
        }
    }
}

// Layer 3: 512 -> 1 dot product, one wave per node; fused has_h mask.
__global__ void final_dot(const unsigned short* __restrict__ H2,
                          const float* __restrict__ W3,
                          const float* __restrict__ b3,
                          const float* __restrict__ cnt,
                          float* __restrict__ out, int col) {
    int wave = (blockIdx.x * blockDim.x + threadIdx.x) >> 5;
    int lane = threadIdx.x & 31;
    if (wave >= N_NODES) return;
    const unsigned short* row = H2 + (long)wave * 512;
    float acc = 0.0f;
#pragma unroll
    for (int t = 0; t < 16; ++t) {
        int j = lane + t * 32;
        acc += bf2f(row[j]) * W3[j];
    }
#pragma unroll
    for (int off = 16; off; off >>= 1)
        acc += __shfl_xor(acc, off, 32);
    if (lane == 0) {
        float mask = (cnt[wave] > 0.0f) ? 1.0f : 0.0f;
        out[(long)wave * 2 + col] = (acc + b3[0]) * mask;
    }
}

// ---------------------------------------------------------------- launcher
extern "C" void kernel_launch(void* const* d_in, const int* in_sizes, int n_in,
                              void* d_out, int out_size, void* d_ws, size_t ws_size,
                              hipStream_t stream) {
    (void)in_sizes; (void)n_in; (void)out_size; (void)ws_size;

    const float* x        = (const float*)d_in[0];
    const int*   z        = (const int*)  d_in[1];
    const int*   batch    = (const int*)  d_in[2];
    const int*   ei       = (const int*)  d_in[3];   // [2,E]
    const int*   solcls   = (const int*)  d_in[4];
    const float* c_emb    = (const float*)d_in[5];
    const float* h_emb    = (const float*)d_in[6];
    const float* cW1 = (const float*)d_in[7];  const float* cb1 = (const float*)d_in[8];
    const float* cW2 = (const float*)d_in[9];  const float* cb2 = (const float*)d_in[10];
    const float* cW3 = (const float*)d_in[11]; const float* cb3 = (const float*)d_in[12];
    const float* hW1 = (const float*)d_in[13]; const float* hb1 = (const float*)d_in[14];
    const float* hW2 = (const float*)d_in[15]; const float* hb2 = (const float*)d_in[16];
    const float* hW3 = (const float*)d_in[17]; const float* hb3 = (const float*)d_in[18];
    float* out = (float*)d_out;

    const int* src = ei;
    const int* dst = ei + N_EDGES;

    // Workspace carve-out (~1.16 GB), 256B aligned slices.
    char* ws = (char*)d_ws;
    size_t off = 0;
    auto alloc = [&](size_t bytes) -> char* {
        char* p = ws + off;
        off += (bytes + 255) & ~(size_t)255;
        return p;
    };
    float*          h_sum    = (float*)         alloc((size_t)N_NODES * DIN * 4);
    float*          cnt      = (float*)         alloc((size_t)N_NODES * 4);
    int*            sol_node = (int*)           alloc((size_t)N_NODES * 4);
    unsigned short* A_c      = (unsigned short*)alloc((size_t)N_NODES * DIN * 2);
    unsigned short* A_h      = (unsigned short*)alloc((size_t)N_NODES * DIN * 2);
    unsigned short* H1       = (unsigned short*)alloc((size_t)N_NODES * 256 * 2);
    unsigned short* H2       = (unsigned short*)alloc((size_t)N_NODES * 512 * 2);
    unsigned short* B1       = (unsigned short*)alloc((size_t)DIN * 256 * 2);
    unsigned short* B2       = (unsigned short*)alloc((size_t)256 * 512 * 2);

    const int mtiles    = N_NODES / 16;        // 18750, exact
    const int mt_groups = mtiles / MT;         // 6250, exact
    const int ng1 = (256 / 16) / NT;           // 4 N-groups (layer 1)
    const int ng2 = (512 / 16) / NT;           // 8 N-groups (layer 2)
    const int blocks1 = mt_groups * ng1 / 8;   // 3125  (2 M-groups/block)
    const int blocks2 = mt_groups * ng2 / 8;   // 6250  (1 M-group/block)
    const int KP1 = DIN + LDS_PAD, KP2 = 256 + LDS_PAD;
    const size_t lds1 = (size_t)(8 / ng1) * (MT * 16) * KP1 * 2;  // 96 x 296 bf16
    const size_t lds2 = (size_t)(8 / ng2) * (MT * 16) * KP2 * 2;  // 48 x 264 bf16

    // Phase 1: graph aggregation
    zero_f32<<<4096, 256, 0, stream>>>(h_sum, (long)N_NODES * DIN);
    zero_f32<<<512,  256, 0, stream>>>(cnt, (long)N_NODES);
    build_feats<<<N_NODES, DIN, 0, stream>>>(x, batch, solcls, c_emb, A_c, sol_node);
    edge_aggregate<<<(N_EDGES * 32 + 255) / 256, 256, 0, stream>>>(
        src, dst, z, sol_node, x, h_emb, h_sum, cnt);
    finalize_havg<<<N_NODES, DIN, 0, stream>>>(h_sum, cnt, A_h);

    // Phase 2: c-path MLP (WMMA bf16, LDS-staged A panels)
    pack_weights<<<(DIN * 256 + 255) / 256, 256, 0, stream>>>(cW1, B1, DIN, 256);
    pack_weights<<<(256 * 512 + 255) / 256, 256, 0, stream>>>(cW2, B2, 256, 512);
    wmma_gemm<<<blocks1, 256, lds1, stream>>>(A_c, B1, cb1, H1, DIN, KP1, 256, ng1, 0);
    wmma_gemm<<<blocks2, 256, lds2, stream>>>(H1,  B2, cb2, H2, 256, KP2, 512, ng2, 1);
    final_dot<<<(N_NODES + 7) / 8, 256, 0, stream>>>(H2, cW3, cb3, cnt, out, 0);

    // Phase 3: h-path MLP (reuses B1/B2/H1/H2 — stream order serializes)
    pack_weights<<<(DIN * 256 + 255) / 256, 256, 0, stream>>>(hW1, B1, DIN, 256);
    pack_weights<<<(256 * 512 + 255) / 256, 256, 0, stream>>>(hW2, B2, 256, 512);
    wmma_gemm<<<blocks1, 256, lds1, stream>>>(A_h, B1, hb1, H1, DIN, KP1, 256, ng1, 0);
    wmma_gemm<<<blocks2, 256, lds2, stream>>>(H1,  B2, hb2, H2, 256, KP2, 512, ng2, 1);
    final_dot<<<(N_NODES + 7) / 8, 256, 0, stream>>>(H2, hW3, hb3, cnt, out, 1);
}
